// MambaPlusBlock_78039555769088
// MI455X (gfx1250) — matrix operand order repaired and checked by reading, not compiled, not run on hardware
//
#include <hip/hip_runtime.h>
#include <hip/hip_bf16.h>

typedef __attribute__((ext_vector_type(16))) __bf16 v16bf;
typedef __attribute__((ext_vector_type(8)))  __bf16 v8bf;
typedef __attribute__((ext_vector_type(8)))  float  v8f;
typedef __attribute__((ext_vector_type(4)))  unsigned int v4u;
typedef __attribute__((ext_vector_type(4)))  int v4i;
typedef __attribute__((ext_vector_type(8)))  int v8i;

#ifndef __has_builtin
#define __has_builtin(x) 0
#endif
#if __has_builtin(__builtin_amdgcn_tensor_load_to_lds) && \
    __has_builtin(__builtin_amdgcn_s_wait_tensorcnt)
#define HAVE_TDM 1
#else
#define HAVE_TDM 0
#endif

#define BATCH 2
#define SEQ   2048
#define DMODEL 512
#define DINNER 1024
#define DSTATE 32
#define DTRANK 64
#define NTOK  (BATCH * SEQ)      // 4096

#define LDS_STRIDE 40            // bf16 elems; 80B rows, 16B aligned
#define CONCAT16(lo, hi) __builtin_shufflevector(lo, hi, 0, 1, 2, 3, 4, 5, 6, 7, \
                                                 8, 9, 10, 11, 12, 13, 14, 15)

__device__ __forceinline__ float sigmoidf_(float x) { return 1.f / (1.f + __expf(-x)); }
__device__ __forceinline__ float siluf_(float x)    { return x / (1.f + __expf(-x)); }

// ---------------------------------------------------------------------------
// Weight convert + transpose: src (K,N) f32 row-major -> dst (N,K) bf16
// ---------------------------------------------------------------------------
__global__ void transpose_bf16_kernel(const float* __restrict__ src,
                                      __bf16* __restrict__ dst, int K, int N) {
  int idx = blockIdx.x * blockDim.x + threadIdx.x;
  if (idx >= K * N) return;
  int n = idx / K;
  int k = idx - n * K;
  dst[idx] = (__bf16)src[(size_t)k * N + n];
}

// ---------------------------------------------------------------------------
// Stage 1: grouped conv3 (512->1024, groups=512) + silu + LayerNorm(1024)
// ---------------------------------------------------------------------------
__global__ __launch_bounds__(256)
void conv1_ln_kernel(const float* __restrict__ x, const float* __restrict__ cw,
                     const float* __restrict__ cb, const float* __restrict__ g2,
                     const float* __restrict__ b2, float* __restrict__ xc,
                     __bf16* __restrict__ xc_bf, __bf16* __restrict__ x_bf) {
  const int m = blockIdx.x;            // token index 0..4095
  const int l = m & (SEQ - 1);
  const int tid = threadIdx.x;
  __shared__ float rs[8], rss[8];
  const float* xrow = x + (size_t)m * DMODEL;

  for (int c = tid; c < DMODEL; c += 256)
    x_bf[(size_t)m * DMODEL + c] = (__bf16)xrow[c];

  float loc[4];
#pragma unroll
  for (int i = 0; i < 4; ++i) {
    int o = tid + i * 256;             // output channel 0..1023
    int c = o >> 1;                    // input channel (2 outputs per group)
    float xm = (l > 0)       ? x[(size_t)(m - 1) * DMODEL + c] : 0.f;
    float x0 = xrow[c];
    float xp = (l < SEQ - 1) ? x[(size_t)(m + 1) * DMODEL + c] : 0.f;
    float a = xm * cw[o * 3 + 0] + x0 * cw[o * 3 + 1] + xp * cw[o * 3 + 2] + cb[o];
    loc[i] = siluf_(a);
  }
  float s = 0.f, ss = 0.f;
#pragma unroll
  for (int i = 0; i < 4; ++i) { s += loc[i]; ss += loc[i] * loc[i]; }
#pragma unroll
  for (int off = 16; off; off >>= 1) {
    s  += __shfl_xor(s,  off, 32);
    ss += __shfl_xor(ss, off, 32);
  }
  if ((tid & 31) == 0) { rs[tid >> 5] = s; rss[tid >> 5] = ss; }
  __syncthreads();
  if (tid == 0) {
    float S = 0.f, SS = 0.f;
    for (int w = 0; w < 8; ++w) { S += rs[w]; SS += rss[w]; }
    float mean = S * (1.f / DINNER);
    rs[0]  = mean;
    rss[0] = SS * (1.f / DINNER) - mean * mean;
  }
  __syncthreads();
  const float mean = rs[0];
  const float inv  = rsqrtf(rss[0] + 1e-5f);
#pragma unroll
  for (int i = 0; i < 4; ++i) {
    int o = tid + i * 256;
    float v = (loc[i] - mean) * inv * g2[o] + b2[o];
    xc[(size_t)m * DINNER + o]    = v;
    xc_bf[(size_t)m * DINNER + o] = (__bf16)v;
  }
}

// ---------------------------------------------------------------------------
// Stage 3: causal depthwise conv4 + silu on xs = xz[:, :1024]
// ---------------------------------------------------------------------------
__global__ __launch_bounds__(256)
void conv2_silu_kernel(const float* __restrict__ xz, const float* __restrict__ w,
                       const float* __restrict__ wb, float* __restrict__ xs_act,
                       __bf16* __restrict__ xs_bf) {
  const int m = blockIdx.x;
  const int b = m >> 11;
  const int l = m & (SEQ - 1);
  for (int d = threadIdx.x; d < DINNER; d += 256) {
    float acc = wb[d];
#pragma unroll
    for (int k = 0; k < 4; ++k) {
      int ll = l - 3 + k;
      if (ll >= 0)
        acc += xz[(size_t)(b * SEQ + ll) * (2 * DINNER) + d] * w[d * 4 + k];
    }
    float v = siluf_(acc);
    xs_act[(size_t)m * DINNER + d] = v;
    xs_bf[(size_t)m * DINNER + d]  = (__bf16)v;
  }
}

// ---------------------------------------------------------------------------
// TDM: issue a 2D bf16 tile load (32 elems/row x `rows`) global -> LDS,
// with hardware padding +16B after every 64B row => LDS row stride 80B.
// Descriptor packing per CDNA5 ISA section 8.3 / 8.4.
// ---------------------------------------------------------------------------
#if HAVE_TDM
__device__ __forceinline__ void tdm_load_tile(const __bf16* gsrc,
                                              unsigned lds_byte_addr,
                                              int rows, int ld_elems) {
  unsigned long long ga = (unsigned long long)(uintptr_t)gsrc;
  const unsigned TD = 1u << 30;  // huge tensor dims -> no OOB clipping
  v4u g0; v8i g1;
  g0[0] = 1u;                                             // count=1 (valid)
  g0[1] = lds_byte_addr;                                  // lds_addr
  g0[2] = (unsigned)(ga & 0xffffffffu);                   // global_addr lo
  g0[3] = (unsigned)((ga >> 32) & 0x1ffffffu) | (2u << 30); // addr hi | type=2
  g1[0] = (int)((1u << 16)      // data_size = 1 -> 2 bytes
              | (1u << 20)      // pad_enable
              | (3u << 22)      // pad_interval: 16 DWORDs (64B)
              | (3u << 25));    // pad_amount:   4 DWORDs (16B)
  g1[1] = (int)((TD & 0xffffu) << 16);                    // tensor_dim0 lo16
  g1[2] = (int)(((TD >> 16) & 0xffffu) | ((TD & 0xffffu) << 16));
  g1[3] = (int)(((TD >> 16) & 0xffffu) | (32u << 16));    // tile_dim0 = 32
  g1[4] = (int)((unsigned)rows & 0xffffu);                // tile_dim1 = rows
  g1[5] = (int)(unsigned)ld_elems;                        // dim0 stride lo32
  g1[6] = 0;                                              // stride hi | s1 lo
  g1[7] = 0;
#if __has_builtin(__builtin_amdgcn_tensor_load_to_lds_d2)
  __builtin_amdgcn_tensor_load_to_lds_d2(g0, g1, 0);
#else
  v4i g2 = {0, 0, 0, 0};
  v4i g3 = {0, 0, 0, 0};
  v8i g4 = {0, 0, 0, 0, 0, 0, 0, 0};
  __builtin_amdgcn_tensor_load_to_lds(g0, g1, g2, g3, g4, 0);
#endif
}
#endif

// ---------------------------------------------------------------------------
// bf16 WMMA GEMM:  C = epi( A[M,K] * B[K,N] ), B given as Bt[N,K].
// Block tile 64(M) x 128(N), 8 waves (4x2); each wave = 16x64 via 4 WMMAs
// sharing one A fragment.  A/B tiles staged to LDS by the Tensor Data Mover
// (double buffered, TENSORcnt) when available, else vector loads.
// ---------------------------------------------------------------------------
enum { EPI_F32 = 0, EPI_F32_BF16 = 1, EPI_SOFTPLUS = 2, EPI_SIGMOID = 3,
       EPI_COMBINE = 4, EPI_BIAS_RES = 5 };

template <int EPI>
__device__ __forceinline__ void epi_store(float v, int row, int col,
                                          float* Cf, __bf16* Cb, int ldc,
                                          const float* bias, const float* aux0,
                                          const float* aux1, int aux_ld) {
  size_t idx = (size_t)row * ldc + col;
  if constexpr (EPI == EPI_F32) {
    Cf[idx] = v;
  } else if constexpr (EPI == EPI_F32_BF16) {
    Cf[idx] = v;
    Cb[idx] = (__bf16)v;
  } else if constexpr (EPI == EPI_SOFTPLUS) {
    float t = v + bias[col];
    Cf[idx] = (t > 20.f) ? t : log1pf(__expf(t));
  } else if constexpr (EPI == EPI_SIGMOID) {
    float t = v + bias[col];
    Cf[idx] = sigmoidf_(t);
  } else if constexpr (EPI == EPI_COMBINE) {
    float g = aux0[(size_t)row * aux_ld + (col >> 1)];
    float out = v * siluf_(g) + aux1[(size_t)row * ldc + col] * (1.f - g);
    Cb[idx] = (__bf16)out;
  } else if constexpr (EPI == EPI_BIAS_RES) {
    Cf[idx] = v + bias[col] + aux0[(size_t)row * aux_ld + col];
  }
}

__device__ __forceinline__ void wmma_compute(const __bf16* As, const __bf16* Bs,
                                             int wm, int wn, int hi, int lm,
                                             v8f acc[4]) {
  const __bf16* ap = &As[(wm * 16 + lm) * LDS_STRIDE + hi * 16];
  v8bf alo = *(const v8bf*)ap;
  v8bf ahi = *(const v8bf*)(ap + 8);
  v16bf afrag = CONCAT16(alo, ahi);
  const __bf16* bbase = &Bs[(wn * 64 + lm) * LDS_STRIDE + hi * 16];
#pragma unroll
  for (int t = 0; t < 4; ++t) {
    const __bf16* bp = bbase + t * 16 * LDS_STRIDE;
    v16bf bfrag = CONCAT16(*(const v8bf*)bp, *(const v8bf*)(bp + 8));
    acc[t] = __builtin_amdgcn_wmma_f32_16x16x32_bf16(false, afrag, false, bfrag,
                                                     (short)0, acc[t], false, false);
  }
}

template <int EPI>
__global__ __launch_bounds__(256)
void gemm_bf16_wmma(const __bf16* __restrict__ A, int lda,
                    const __bf16* __restrict__ Bt, int ldb,
                    int M, int N, int K,
                    float* __restrict__ Cf, __bf16* __restrict__ Cb, int ldc,
                    const float* __restrict__ bias,
                    const float* __restrict__ aux0,
                    const float* __restrict__ aux1, int aux_ld) {
  __shared__ __align__(16) __bf16 As[2][64 * LDS_STRIDE];
  __shared__ __align__(16) __bf16 Bs[2][128 * LDS_STRIDE];

  const int tid  = threadIdx.x;
  const int wave = tid >> 5;
  const int lane = tid & 31;
  const int wm = wave >> 1, wn = wave & 1;   // 4x2 wave grid
  const int hi = lane >> 4, lm = lane & 15;
  const int m0 = blockIdx.y * 64;
  const int n0 = blockIdx.x * 128;

  v8f acc[4] = {v8f{}, v8f{}, v8f{}, v8f{}};

#if HAVE_TDM
  const __bf16* gA = A  + (size_t)m0 * lda;
  const __bf16* gB = Bt + (size_t)n0 * ldb;
  if (wave == 0) {
    tdm_load_tile(gA, (unsigned)(uintptr_t)&As[0][0], 64,  lda);
    tdm_load_tile(gB, (unsigned)(uintptr_t)&Bs[0][0], 128, ldb);
  }
  int cur = 0;
  for (int k0 = 0; k0 < K; k0 += 32) {
    const bool more = (k0 + 32) < K;
    if (wave == 0) {
      if (more) {
        tdm_load_tile(gA + k0 + 32, (unsigned)(uintptr_t)&As[cur ^ 1][0], 64,  lda);
        tdm_load_tile(gB + k0 + 32, (unsigned)(uintptr_t)&Bs[cur ^ 1][0], 128, ldb);
        __builtin_amdgcn_s_wait_tensorcnt((short)2);   // current buffer done
      } else {
        __builtin_amdgcn_s_wait_tensorcnt((short)0);   // drain
      }
    }
    __syncthreads();
    wmma_compute(&As[cur][0], &Bs[cur][0], wm, wn, hi, lm, acc);
    __syncthreads();                                   // release buf before refill
    cur ^= 1;
  }
#else
  const int lrow = tid >> 2;        // 0..63
  const int lcol = (tid & 3) * 8;   // 0,8,16,24
  const __bf16* gA  = A  + (size_t)(m0 + lrow) * lda + lcol;
  const __bf16* gB0 = Bt + (size_t)(n0 + lrow) * ldb + lcol;
  const __bf16* gB1 = Bt + (size_t)(n0 + 64 + lrow) * ldb + lcol;
  for (int k0 = 0; k0 < K; k0 += 32) {
    v8bf av  = *(const v8bf*)(gA + k0);
    v8bf bv0 = *(const v8bf*)(gB0 + k0);
    v8bf bv1 = *(const v8bf*)(gB1 + k0);
    if (k0 + 32 < K) {
      __builtin_prefetch(gA + k0 + 32, 0, 1);
      __builtin_prefetch(gB0 + k0 + 32, 0, 1);
      __builtin_prefetch(gB1 + k0 + 32, 0, 1);
    }
    __syncthreads();
    *(v8bf*)(&As[0][lrow * LDS_STRIDE + lcol])        = av;
    *(v8bf*)(&Bs[0][lrow * LDS_STRIDE + lcol])        = bv0;
    *(v8bf*)(&Bs[0][(lrow + 64) * LDS_STRIDE + lcol]) = bv1;
    __syncthreads();
    wmma_compute(&As[0][0], &Bs[0][0], wm, wn, hi, lm, acc);
  }
#endif

  // Epilogue: C VGPR r -> row = 8*hi + r; lane lm -> col
#pragma unroll
  for (int t = 0; t < 4; ++t) {
#pragma unroll
    for (int r = 0; r < 8; ++r) {
      int row = m0 + wm * 16 + 8 * hi + r;
      int col = n0 + wn * 64 + t * 16 + lm;
      epi_store<EPI>(acc[t][r], row, col, Cf, Cb, ldc, bias, aux0, aux1, aux_ld);
    }
  }
}

// ---------------------------------------------------------------------------
// Stage 5: selective-scan. One wave per (batch, channel); lane = state index.
// ---------------------------------------------------------------------------
__global__ __launch_bounds__(256)
void scan_kernel(const float* __restrict__ delta, const float* __restrict__ dbl,
                 const float* __restrict__ xs_act, const float* __restrict__ xz,
                 const float* __restrict__ A_log, const float* __restrict__ Dp,
                 __bf16* __restrict__ yact) {
  const int wave = (blockIdx.x * 256 + threadIdx.x) >> 5;  // 0..2047
  const int lane = threadIdx.x & 31;
  const int b = wave >> 10;
  const int d = wave & (DINNER - 1);

  const float An = -__expf(A_log[d * DSTATE + lane]);
  const float Dv = Dp[d];
  float h = 0.f;
  const size_t base = (size_t)b * SEQ;

  for (int t = 0; t < SEQ; ++t) {
    const size_t m = base + t;
    const float dt = delta[m * DINNER + d];
    const float u  = xs_act[m * DINNER + d];
    const float Bt = dbl[m * 128 + DTRANK + lane];
    const float Ct = dbl[m * 128 + DTRANK + DSTATE + lane];
    const float dA = __expf(dt * An);
    h = dA * h + (dt * u) * Bt;
    float contrib = h * Ct;
#pragma unroll
    for (int off = 16; off; off >>= 1) contrib += __shfl_xor(contrib, off, 32);
    if (lane == 0) {
      float y  = contrib + u * Dv;
      float zv = xz[m * (2 * DINNER) + DINNER + d];
      yact[m * DINNER + d] = (__bf16)(y * siluf_(zv));
    }
  }
}

// ---------------------------------------------------------------------------
// Stage 7: final LayerNorm(512)
// ---------------------------------------------------------------------------
__global__ __launch_bounds__(256)
void ln_final_kernel(const float* __restrict__ pre, const float* __restrict__ g,
                     const float* __restrict__ bb, float* __restrict__ out) {
  const int m = blockIdx.x;
  const int tid = threadIdx.x;
  __shared__ float rs[8], rss[8];
  float v0 = pre[(size_t)m * DMODEL + tid];
  float v1 = pre[(size_t)m * DMODEL + 256 + tid];
  float s = v0 + v1, ss = v0 * v0 + v1 * v1;
#pragma unroll
  for (int off = 16; off; off >>= 1) {
    s  += __shfl_xor(s,  off, 32);
    ss += __shfl_xor(ss, off, 32);
  }
  if ((tid & 31) == 0) { rs[tid >> 5] = s; rss[tid >> 5] = ss; }
  __syncthreads();
  if (tid == 0) {
    float S = 0.f, SS = 0.f;
    for (int w = 0; w < 8; ++w) { S += rs[w]; SS += rss[w]; }
    float mean = S * (1.f / DMODEL);
    rs[0]  = mean;
    rss[0] = SS * (1.f / DMODEL) - mean * mean;
  }
  __syncthreads();
  const float mean = rs[0];
  const float inv  = rsqrtf(rss[0] + 1e-5f);
  out[(size_t)m * DMODEL + tid]       = (v0 - mean) * inv * g[tid] + bb[tid];
  out[(size_t)m * DMODEL + 256 + tid] = (v1 - mean) * inv * g[256 + tid] + bb[256 + tid];
}

// ---------------------------------------------------------------------------
extern "C" void kernel_launch(void* const* d_in, const int* in_sizes, int n_in,
                              void* d_out, int out_size, void* d_ws, size_t ws_size,
                              hipStream_t stream) {
  const float* x          = (const float*)d_in[0];
  const float* conv_w     = (const float*)d_in[1];
  const float* conv_b     = (const float*)d_in[2];
  const float* gate_w     = (const float*)d_in[3];
  const float* gate_b     = (const float*)d_in[4];
  const float* out_w      = (const float*)d_in[5];
  const float* out_b      = (const float*)d_in[6];
  const float* ln1_g      = (const float*)d_in[7];
  const float* ln1_b      = (const float*)d_in[8];
  const float* ln2_g      = (const float*)d_in[9];
  const float* ln2_b      = (const float*)d_in[10];
  const float* in_proj_w  = (const float*)d_in[11];
  const float* ssm_conv_w = (const float*)d_in[12];
  const float* ssm_conv_b = (const float*)d_in[13];
  const float* x_proj_w   = (const float*)d_in[14];
  const float* dt_proj_w  = (const float*)d_in[15];
  const float* dt_proj_b  = (const float*)d_in[16];
  const float* A_log      = (const float*)d_in[17];
  const float* D_param    = (const float*)d_in[18];
  const float* out_proj_w = (const float*)d_in[19];
  float* out = (float*)d_out;

  char* ws = (char*)d_ws;
  size_t off = 0;
  auto alloc = [&](size_t bytes) {
    size_t o = off;
    off = (off + bytes + 255) & ~(size_t)255;
    return o;
  };
  __bf16* WinT  = (__bf16*)(ws + alloc((size_t)2048 * 1024 * 2));
  __bf16* WgT   = (__bf16*)(ws + alloc((size_t)512 * 512 * 2));
  __bf16* WxT   = (__bf16*)(ws + alloc((size_t)128 * 1024 * 2));
  __bf16* WdtT  = (__bf16*)(ws + alloc((size_t)1024 * 64 * 2));
  __bf16* WopT  = (__bf16*)(ws + alloc((size_t)1024 * 1024 * 2));
  __bf16* WoT   = (__bf16*)(ws + alloc((size_t)512 * 1024 * 2));
  __bf16* x_bf  = (__bf16*)(ws + alloc((size_t)NTOK * DMODEL * 2));
  float*  xc    = (float*) (ws + alloc((size_t)NTOK * DINNER * 4));
  __bf16* xc_bf = (__bf16*)(ws + alloc((size_t)NTOK * DINNER * 2));
  float*  xz    = (float*) (ws + alloc((size_t)NTOK * 2 * DINNER * 4));
  float*  xs_a  = (float*) (ws + alloc((size_t)NTOK * DINNER * 4));
  __bf16* xs_bf = (__bf16*)(ws + alloc((size_t)NTOK * DINNER * 2));
  float*  dbl   = (float*) (ws + alloc((size_t)NTOK * 128 * 4));
  __bf16* dblbf = (__bf16*)(ws + alloc((size_t)NTOK * 128 * 2));
  float*  delta = (float*) (ws + alloc((size_t)NTOK * DINNER * 4));
  float*  gate  = (float*) (ws + alloc((size_t)NTOK * DMODEL * 4));
  __bf16* yact  = (__bf16*)(ws + alloc((size_t)NTOK * DINNER * 2));
  __bf16* comb  = (__bf16*)(ws + alloc((size_t)NTOK * DINNER * 2));
  float*  pre   = (float*) (ws + alloc((size_t)NTOK * DMODEL * 4));
  (void)ws_size; (void)in_sizes; (void)n_in; (void)out_size;

  dim3 blk(256);
  auto tgrid = [](int elems) { return dim3((elems + 255) / 256); };

  transpose_bf16_kernel<<<tgrid(1024 * 2048), blk, 0, stream>>>(in_proj_w,  WinT, 1024, 2048);
  transpose_bf16_kernel<<<tgrid(512 * 512),   blk, 0, stream>>>(gate_w,     WgT,  512,  512);
  transpose_bf16_kernel<<<tgrid(1024 * 128),  blk, 0, stream>>>(x_proj_w,   WxT,  1024, 128);
  transpose_bf16_kernel<<<tgrid(64 * 1024),   blk, 0, stream>>>(dt_proj_w,  WdtT, 64,   1024);
  transpose_bf16_kernel<<<tgrid(1024 * 1024), blk, 0, stream>>>(out_proj_w, WopT, 1024, 1024);
  transpose_bf16_kernel<<<tgrid(1024 * 512),  blk, 0, stream>>>(out_w,      WoT,  1024, 512);

  conv1_ln_kernel<<<dim3(NTOK), blk, 0, stream>>>(x, conv_w, conv_b, ln2_g, ln2_b,
                                                  xc, xc_bf, x_bf);

  // xz = xc @ in_proj_w   [4096,1024]x[1024,2048]
  gemm_bf16_wmma<EPI_F32><<<dim3(2048 / 128, NTOK / 64), blk, 0, stream>>>(
      xc_bf, 1024, WinT, 1024, NTOK, 2048, 1024, xz, nullptr, 2048,
      nullptr, nullptr, nullptr, 0);

  conv2_silu_kernel<<<dim3(NTOK), blk, 0, stream>>>(xz, ssm_conv_w, ssm_conv_b,
                                                    xs_a, xs_bf);

  // dbl = xs @ x_proj_w  [4096,1024]x[1024,128]
  gemm_bf16_wmma<EPI_F32_BF16><<<dim3(128 / 128, NTOK / 64), blk, 0, stream>>>(
      xs_bf, 1024, WxT, 1024, NTOK, 128, 1024, dbl, dblbf, 128,
      nullptr, nullptr, nullptr, 0);

  // delta = softplus(dt @ dt_proj_w + dt_proj_b)   K=64
  gemm_bf16_wmma<EPI_SOFTPLUS><<<dim3(1024 / 128, NTOK / 64), blk, 0, stream>>>(
      dblbf, 128, WdtT, 64, NTOK, 1024, 64, delta, nullptr, 1024,
      dt_proj_b, nullptr, nullptr, 0);

  // gate = sigmoid(x @ gate_w + gate_b)
  gemm_bf16_wmma<EPI_SIGMOID><<<dim3(512 / 128, NTOK / 64), blk, 0, stream>>>(
      x_bf, 512, WgT, 512, NTOK, 512, 512, gate, nullptr, 512,
      gate_b, nullptr, nullptr, 0);

  scan_kernel<<<dim3(BATCH * DINNER / 8), blk, 0, stream>>>(
      delta, dbl, xs_a, xz, A_log, D_param, yact);

  // combined = (yact @ out_proj_w)*silu(ge) + xc*(1-ge)
  gemm_bf16_wmma<EPI_COMBINE><<<dim3(1024 / 128, NTOK / 64), blk, 0, stream>>>(
      yact, 1024, WopT, 1024, NTOK, 1024, 1024, nullptr, comb, 1024,
      nullptr, gate, xc, 512);

  // pre = combined @ out_w + out_b + residual(x)
  gemm_bf16_wmma<EPI_BIAS_RES><<<dim3(512 / 128, NTOK / 64), blk, 0, stream>>>(
      comb, 1024, WoT, 1024, NTOK, 512, 1024, pre, nullptr, 512,
      out_b, x, nullptr, 512);

  ln_final_kernel<<<dim3(NTOK), blk, 0, stream>>>(pre, ln1_g, ln1_b, out);
}
